// rnnode_70308614636160
// MI455X (gfx1250) — matrix operand order repaired and checked by reading, not compiled, not run on hardware
//
#include <hip/hip_runtime.h>
#include <math.h>
#include <stdint.h>

// Problem sizes (fixed by the reference)
#define B_   128
#define T_   512
#define I_   128
#define H_   512
#define O_   64
#define FP_ITERS 64
#define KCH  20          // (I_+H_)/32 K-chunks of the fused [u|x] GEMM

typedef __bf16  bf16_t;
typedef bf16_t  v16bf __attribute__((ext_vector_type(16)));
typedef float   v8f   __attribute__((ext_vector_type(8)));
typedef int     v4i   __attribute__((ext_vector_type(4)));
typedef unsigned int u32x4 __attribute__((ext_vector_type(4)));
typedef int     i32x8 __attribute__((ext_vector_type(8)));
typedef int     i32x4 __attribute__((ext_vector_type(4)));

union Frag { v16bf bf; v4i i2[2]; };

__device__ __forceinline__ unsigned int rfl(unsigned int v) {
  return (unsigned int)__builtin_amdgcn_readfirstlane((int)v);
}

__device__ __forceinline__ unsigned short f2bf(float f) {
  unsigned int u = __float_as_uint(f);
  u += 0x7FFFu + ((u >> 16) & 1u);          // round-to-nearest-even
  return (unsigned short)(u >> 16);
}

// ---------------------------------------------------------------------------
// Kernel 1a: pre-swizzle weights into WMMA B-fragment layout (bf16).
//   wcat: [32 ntiles][20 kchunks][32 lanes][16 elems]  = [W_ih | W_hh]^T tiles
//   wout: [ 4 ntiles][16 kchunks][32 lanes][16 elems]  = W_out^T tiles
// B-matrix (32x16 bf16) lane layout: lane = N + 16*khi, elems e hold
// K = kchunk*32 + khi*16 + e  (contiguous 16 K per lane -> 32B per lane).
// ---------------------------------------------------------------------------
__global__ void prep_weights(const float* __restrict__ W_ih,
                             const float* __restrict__ W_hh,
                             const float* __restrict__ W_out,
                             unsigned short* __restrict__ wcat,
                             unsigned short* __restrict__ wout) {
  int idx = blockIdx.x * blockDim.x + threadIdx.x;
  const int total_wcat = 32 * KCH * 32 * 16;   // 327680
  const int total_wout = 4 * 16 * 32 * 16;     // 32768
  if (idx < total_wcat) {
    int e    = idx & 15;
    int lane = (idx >> 4) & 31;
    int kc   = (idx >> 9) % KCH;
    int nt   = (idx >> 9) / KCH;
    int n    = nt * 16 + (lane & 15);
    int k    = kc * 32 + (lane >> 4) * 16 + e;
    float v  = (k < I_) ? W_ih[n * I_ + k] : W_hh[n * H_ + (k - I_)];
    wcat[idx] = f2bf(v);
  } else if (idx < total_wcat + total_wout) {
    int j    = idx - total_wcat;
    int e    = j & 15;
    int lane = (j >> 4) & 31;
    int kc   = (j >> 9) & 15;
    int nt   = (j >> 9) >> 4;
    int n    = nt * 16 + (lane & 15);
    int k    = kc * 32 + (lane >> 4) * 16 + e;
    wout[j] = f2bf(W_out[n * H_ + k]);
  }
}

// Kernel 1b: one-shot f32 -> bf16 conversion of the whole input tensor so the
// per-step staging in the recurrent kernel is a raw TDM byte copy.
__global__ void prep_u(const float* __restrict__ u,
                       unsigned short* __restrict__ ubf) {
  int i = blockIdx.x * blockDim.x + threadIdx.x;
  if (i < B_ * T_ * I_) ubf[i] = f2bf(u[i]);
}

// ---------------------------------------------------------------------------
// TDM: load one 16x128 bf16 tile (batch rows m0..m0+15 at timestep t, row
// stride T*I elements) from global into a contiguous 4KB LDS buffer.
// D# per CDNA5 ISA (2D tensor, groups 2/3 zero).
// All variable words are pre-readfirstlane'd by the caller so every element
// of the descriptor vectors is SGPR-resident (intrinsic needs SGPR groups).
// ---------------------------------------------------------------------------
__device__ __forceinline__ void tdm_load_u(unsigned int ga_lo,
                                           unsigned int ga_hi,
                                           unsigned int lds_off) {
  u32x4 g0;
  g0[0] = 1u;                                  // count=1, user descriptor
  g0[1] = lds_off;                             // lds_addr (bytes)
  g0[2] = ga_lo;                               // global_addr[31:0]
  g0[3] = (ga_hi & 0x01FFFFFFu) | 0x80000000u; // global_addr[56:32], type=2
  i32x8 g1;
  g1[0] = 0x00010000;            // workgroup_mask=0, data_size=1 (2 bytes)
  g1[1] = (int)(128u << 16);     // tensor_dim0 = 128 (I)
  g1[2] = (int)(16u << 16);      // tensor_dim1 = 16  (rows)
  g1[3] = (int)(128u << 16);     // tile_dim0 = 128
  g1[4] = 16;                    // tile_dim1 = 16, tile_dim2 = 0
  g1[5] = (int)(T_ * I_);        // tensor_dim0_stride = 65536 elements
  g1[6] = 0;
  g1[7] = 0;
  i32x4 z4 = {0, 0, 0, 0};
  i32x8 z8 = {0, 0, 0, 0, 0, 0, 0, 0};
  __builtin_amdgcn_tensor_load_to_lds(g0, g1, z4, z4, z8, 0);
}

// ---------------------------------------------------------------------------
// Fused GEMM over the K=640 concatenated [u_t | x] operand held in LDS.
// A-matrix (16x32 bf16) lane layout (contiguous k, same order as wcat): lane = M + 16*khi,
//   VGPR0-3 hold K = kc*32 + khi*16 + {0..7}, VGPR4-7 hold K = +8.
// The opaque zero OFFSET (not an opaque pointer) defeats LICM so the 20
// B-fragment loads stay inside the timestep loop (L2 hits, no spills) while
// preserving the pointer's global address-space provenance so the loads lower
// to global_load_b128 rather than flat_load_b128.
// ---------------------------------------------------------------------------
__device__ __forceinline__ v8f gemm640(const unsigned short* ush,
                                       const unsigned short* xsh,
                                       const unsigned short* __restrict__ wfrag,
                                       int lane, int khi) {
  unsigned int opq = 0;
  asm volatile("" : "+s"(opq));
  const unsigned short* wf = wfrag + opq;

  v8f acc;
  #pragma unroll
  for (int i = 0; i < 8; ++i) acc[i] = 0.f;
  const int row = lane & 15;
  #pragma unroll
  for (int kc = 0; kc < KCH; ++kc) {
    Frag a, b;
    const unsigned short* ap = (kc < 4)
        ? (ush + row * I_ + kc * 32 + khi * 16)
        : (xsh + row * H_ + (kc - 4) * 32 + khi * 16);
    a.i2[0] = *(const v4i*)(ap);          // K = kc*32 + khi*16 + 0..7
    a.i2[1] = *(const v4i*)(ap + 8);      // K = kc*32 + khi*16 + 8..15
    const unsigned short* bp = wf + kc * 512 + lane * 16;
    b.i2[0] = *(const v4i*)(bp);
    b.i2[1] = *(const v4i*)(bp + 8);
    acc = __builtin_amdgcn_wmma_f32_16x16x32_bf16(false, a.bf, false, b.bf,
                                                  (short)0, acc, false, false);
  }
  return acc;
}

// ---------------------------------------------------------------------------
// Kernel 2: persistent recurrent kernel. 8 workgroups (16 batch rows each),
// 1024 threads = 32 waves, wave w owns hidden columns [16w, 16w+16).
// State x kept in f32 registers (C-tile layout) + bf16 mirror in LDS.
// u_t staged by the TDM, double-buffered to overlap the DMA with the WMMAs.
// ---------------------------------------------------------------------------
__global__ __launch_bounds__(1024) void rnn_recurrent(
    const unsigned short* __restrict__ ubf,    // [B,T,I] bf16
    const float* __restrict__ b_ih,
    const float* __restrict__ b_hh,
    const float* __restrict__ tau,
    const unsigned short* __restrict__ wcat,   // fragment layout
    unsigned short* __restrict__ outrnn) {     // [B][T][H] bf16 trace
  __shared__ __attribute__((aligned(16))) unsigned short xsh[16 * H_];     // 16KB
  __shared__ __attribute__((aligned(16))) unsigned short ush[2][16 * I_];  // 2x4KB

  const int tid  = threadIdx.x;
  const int lane = tid & 31;
  const int w    = tid >> 5;           // wave id == N-tile
  const int m0   = blockIdx.x * 16;    // batch row base
  const int khi  = lane >> 4;
  const int n    = w * 16 + (lane & 15);

  const float bias = b_ih[n] + b_hh[n];
  const float tn   = tau[n];
  const float et   = expf(tn);
  const float iet  = expf(-tn);
  const unsigned short* wfrag = wcat + w * (KCH * 512);

  // LDS byte offsets of the two u buffers, forced uniform/SGPR
  const unsigned int uoff0 = rfl((unsigned int)(uintptr_t)(&ush[0][0]));
  const unsigned int uoff1 = rfl((unsigned int)(uintptr_t)(&ush[1][0]));
  // global base of this WG's batch-row slice of u (bytes), forced uniform
  unsigned long long gbase = (unsigned long long)(uintptr_t)(ubf + (size_t)m0 * T_ * I_);
  const unsigned int gb_lo = rfl((unsigned int)gbase);
  const unsigned int gb_hi = rfl((unsigned int)(gbase >> 32));

  // prologue: TDM-stage u_0 into buffer 0 while everyone zeroes x
  if (w == 0) tdm_load_u(gb_lo, gb_hi, uoff0);
  for (int i = tid; i < 16 * H_; i += 1024) xsh[i] = 0;
  if (w == 0) __builtin_amdgcn_s_wait_tensorcnt(0);
  __syncthreads();

  v8f xv;
  #pragma unroll
  for (int i = 0; i < 8; ++i) xv[i] = 0.f;

  // --- fixed-point phase: x = exp(tau)*tanh([u0|x] @ Wcat^T + bias) ---
  for (int it = 0; it < FP_ITERS; ++it) {
    v8f acc = gemm640(ush[0], xsh, wfrag, lane, khi);
    #pragma unroll
    for (int i = 0; i < 8; ++i) xv[i] = et * tanhf(acc[i] + bias);
    __syncthreads();                          // everyone done reading old x
    #pragma unroll
    for (int v = 0; v < 8; ++v)
      xsh[(v + 8 * khi) * H_ + n] = f2bf(xv[v]);
    __syncthreads();                          // new x visible
  }
  { // store x0 as outrnn[:, t=0, :]
    int r = tid >> 6, ck = tid & 63;
    v4i d = *(const v4i*)(xsh + r * H_ + ck * 8);
    *(v4i*)(outrnn + (((size_t)(m0 + r)) * T_ + 0) * H_ + ck * 8) = d;
  }

  // --- Euler phase: x += tanh([u_s|x] @ Wcat^T + bias) - x*exp(-tau) ---
  for (int s = 0; s < T_ - 1; ++s) {
    const int cur = s & 1;
    // prefetch u_{s+1} into the other buffer; overlaps with this step's WMMAs
    if (w == 0 && s < T_ - 2) {
      // byte offset of timestep s+1 within this WG's slice: (s+1)*I*2
      unsigned int off = rfl((unsigned int)((s + 1) * I_ * 2));
      unsigned int lo  = gb_lo + off;
      unsigned int hi  = gb_hi + (lo < gb_lo ? 1u : 0u);
      tdm_load_u(lo, hi, cur ? uoff0 : uoff1);
    }
    v8f acc = gemm640(cur ? ush[1] : ush[0], xsh, wfrag, lane, khi);
    #pragma unroll
    for (int i = 0; i < 8; ++i)
      xv[i] = xv[i] + tanhf(acc[i] + bias) - xv[i] * iet;
    __syncthreads();                          // reads of x_s and u_s done
    #pragma unroll
    for (int v = 0; v < 8; ++v)
      xsh[(v + 8 * khi) * H_ + n] = f2bf(xv[v]);
    if (w == 0) __builtin_amdgcn_s_wait_tensorcnt(0);  // u_{s+1} landed
    __syncthreads();                          // new x + new u visible
    { // store outrnn[:, t=s+1, :] (16B per thread, coalesced)
      int r = tid >> 6, ck = tid & 63;
      v4i d = *(const v4i*)(xsh + r * H_ + ck * 8);
      *(v4i*)(outrnn + (((size_t)(m0 + r)) * T_ + (s + 1)) * H_ + ck * 8) = d;
    }
  }
}

// ---------------------------------------------------------------------------
// Kernel 3: output projection  out[b,t,o] = outrnn[b,t,:] . W_out[o,:] + b_out
// Rows = B*T = 65536, N = 64, K = 512. One wave per 16x16 tile.
// ---------------------------------------------------------------------------
__global__ __launch_bounds__(512) void out_gemm(
    const unsigned short* __restrict__ outrnn,  // [B*T][H] bf16
    const unsigned short* __restrict__ wout,    // fragment layout
    const float* __restrict__ b_out,
    float* __restrict__ out) {                  // [B*T][O] f32
  const int tid  = threadIdx.x;
  const int lane = tid & 31;
  const int tile = blockIdx.x * 16 + (tid >> 5);
  const int mt   = tile >> 2;
  const int nt   = tile & 3;
  const int khi  = lane >> 4;
  const int o    = nt * 16 + (lane & 15);
  const float bo = b_out[o];

  v8f acc;
  #pragma unroll
  for (int i = 0; i < 8; ++i) acc[i] = 0.f;
  const unsigned short* arow = outrnn + (size_t)(mt * 16 + (lane & 15)) * H_ + khi * 16;
  const unsigned short* bbase = wout + nt * (16 * 512);
  #pragma unroll
  for (int kc = 0; kc < 16; ++kc) {
    Frag a, b;
    const unsigned short* ap = arow + kc * 32;
    a.i2[0] = *(const v4i*)(ap);
    a.i2[1] = *(const v4i*)(ap + 8);
    const unsigned short* bp = bbase + kc * 512 + lane * 16;
    b.i2[0] = *(const v4i*)(bp);
    b.i2[1] = *(const v4i*)(bp + 8);
    acc = __builtin_amdgcn_wmma_f32_16x16x32_bf16(false, a.bf, false, b.bf,
                                                  (short)0, acc, false, false);
  }
  #pragma unroll
  for (int v = 0; v < 8; ++v)
    out[(size_t)(mt * 16 + v + 8 * khi) * O_ + o] = acc[v] + bo;
}

// ---------------------------------------------------------------------------
extern "C" void kernel_launch(void* const* d_in, const int* in_sizes, int n_in,
                              void* d_out, int out_size, void* d_ws, size_t ws_size,
                              hipStream_t stream) {
  (void)in_sizes; (void)n_in; (void)out_size; (void)ws_size;
  const float* u     = (const float*)d_in[0];
  const float* W_ih  = (const float*)d_in[1];
  const float* W_hh  = (const float*)d_in[2];
  const float* b_ih  = (const float*)d_in[3];
  const float* b_hh  = (const float*)d_in[4];
  const float* tau   = (const float*)d_in[5];
  const float* W_out = (const float*)d_in[6];
  const float* b_out = (const float*)d_in[7];
  float* out = (float*)d_out;

  // workspace layout:
  //   [0, 640KB)        wcat fragments
  //   [640KB, 704KB)    wout fragments
  //   [1MB, 17MB)       u converted to bf16
  //   [17MB, 81MB)      outrnn state trace (bf16)
  unsigned short* wcat   = (unsigned short*)d_ws;
  unsigned short* wout   = (unsigned short*)((char*)d_ws + 655360);
  unsigned short* ubf    = (unsigned short*)((char*)d_ws + (1u << 20));
  unsigned short* outrnn = (unsigned short*)((char*)d_ws + (17u << 20));

  prep_weights<<<(360448 + 255) / 256, 256, 0, stream>>>(W_ih, W_hh, W_out, wcat, wout);
  prep_u<<<(B_ * T_ * I_ + 255) / 256, 256, 0, stream>>>(u, ubf);
  rnn_recurrent<<<8, 1024, 0, stream>>>(ubf, b_ih, b_hh, tau, wcat, outrnn);
  out_gemm<<<1024, 512, 0, stream>>>(outrnn, wout, b_out, out);
}
